// Attention_71863392797159
// MI455X (gfx1250) — matrix operand order repaired and checked by reading, not compiled, not run on hardware
//
#include <hip/hip_runtime.h>
#include <hip/hip_bf16.h>
#include <stdint.h>

typedef __attribute__((ext_vector_type(16))) __bf16 v16bf;
typedef __attribute__((ext_vector_type(8)))  __bf16 v8bf;
typedef __attribute__((ext_vector_type(8)))  float  v8f;

#define B_    4
#define S_    2048
#define D_    4096
#define H_    32
#define KV_   8
#define HD_   128
#define NREP_ 4
#define SCALE_ 0.08838834764831845f  // 1/sqrt(128)

// ---------------------------------------------------------------------------
// CDNA5 async DMA: global -> LDS, 16 bytes per lane, tracked by ASYNCcnt.
// LDS byte offset lives in addr[31:0] of the flat pointer (ISA 10.2 aperture).
// ---------------------------------------------------------------------------
__device__ __forceinline__ void async_copy_b128(void* lds, const void* gsrc) {
  uint32_t ldsoff = (uint32_t)(uintptr_t)lds;
  unsigned long long ga = (unsigned long long)(uintptr_t)gsrc;
  asm volatile("global_load_async_to_lds_b128 %0, %1, off"
               :
               : "v"(ldsoff), "v"(ga)
               : "memory");
}
__device__ __forceinline__ void wait_async0() {
  asm volatile("s_wait_asynccnt 0" ::: "memory");
}

__device__ __forceinline__ v8f wmma_bf16(v16bf a, v16bf b, v8f c) {
  return __builtin_amdgcn_wmma_f32_16x16x32_bf16(false, a, false, b, (short)0, c,
                                                 false, false);
}

// A-matrix fragment (16x32 bf16, this lane = row m) from a row pointer.
// Contiguous k order (matches B fragments): element e -> k = half*16+e.
__device__ __forceinline__ v16bf make_afrag(const __bf16* rowp, int half) {
  v8bf lo = *(const v8bf*)(rowp + half * 16);
  v8bf hi = *(const v8bf*)(rowp + half * 16 + 8);
  return __builtin_shufflevector(lo, hi, 0, 1, 2, 3, 4, 5, 6, 7, 8, 9, 10, 11,
                                 12, 13, 14, 15);
}

// ---------------------------------------------------------------------------
// fp32 -> bf16 elementwise (8 elements/thread)
// ---------------------------------------------------------------------------
__global__ __launch_bounds__(256) void conv_bf16_kernel(
    const float* __restrict__ s, __bf16* __restrict__ d) {
  size_t i = ((size_t)blockIdx.x * 256 + threadIdx.x) * 8;
  float4 f0 = *(const float4*)(s + i);
  float4 f1 = *(const float4*)(s + i + 4);
  v8bf t8;
  t8[0] = (__bf16)f0.x; t8[1] = (__bf16)f0.y;
  t8[2] = (__bf16)f0.z; t8[3] = (__bf16)f0.w;
  t8[4] = (__bf16)f1.x; t8[5] = (__bf16)f1.y;
  t8[6] = (__bf16)f1.z; t8[7] = (__bf16)f1.w;
  *(v8bf*)(d + i) = t8;
}

// ---------------------------------------------------------------------------
// transpose + convert: w f32 [K][N] -> wT bf16 [N][K]
// ---------------------------------------------------------------------------
template <int N, int K>
__global__ __launch_bounds__(256) void transpose_conv_kernel(
    const float* __restrict__ w, __bf16* __restrict__ wT) {
  __shared__ float tile[32][33];
  const int n0 = blockIdx.x * 32;
  const int k0 = blockIdx.y * 32;
  const int tx = threadIdx.x & 31;
  const int ty = threadIdx.x >> 5;  // 0..7
#pragma unroll
  for (int i = 0; i < 4; ++i) {
    int kr = ty + i * 8;
    tile[kr][tx] = w[(size_t)(k0 + kr) * N + n0 + tx];
  }
  __syncthreads();
#pragma unroll
  for (int i = 0; i < 4; ++i) {
    int nr = ty + i * 8;
    wT[(size_t)(n0 + nr) * K + k0 + tx] = (__bf16)tile[tx][nr];
  }
}

// ---------------------------------------------------------------------------
// GEMM: C[M,N] f32 = A[M,K] bf16 * Bt[N,K] bf16 (B pre-transposed).
// 256 threads = 8 waves (2x4), block tile 128x256, BK=64, wave tile 64x64.
// Double-buffered async-DMA staging: prefetch tile kt+1 while computing kt.
// ---------------------------------------------------------------------------
template <int N, int K>
__global__ __launch_bounds__(256) void gemm_bf16_kernel(
    const __bf16* __restrict__ A, const __bf16* __restrict__ Bt,
    float* __restrict__ C, int M) {
  __shared__ __bf16 As[2][128 * 64];  // [m][k]
  __shared__ __bf16 Bs[2][256 * 64];  // [n][k]

  const int t = threadIdx.x;
  const int lane = t & 31;
  const int wave = t >> 5;
  const int wm = wave >> 2;  // 0..1 -> 64-row strip
  const int wn = wave & 3;   // 0..3 -> 64-col strip
  const int half = lane >> 4;
  const int n16 = lane & 15;

  const int m0 = blockIdx.y * 128;
  const int n0 = blockIdx.x * 256;

  v8f acc[4][4];
#pragma unroll
  for (int i = 0; i < 4; ++i)
#pragma unroll
    for (int j = 0; j < 4; ++j) acc[i][j] = (v8f){0, 0, 0, 0, 0, 0, 0, 0};

  auto stage = [&](int k0, int buf) {
#pragma unroll
    for (int i = 0; i < 4; ++i) {  // A: 1024 chunks of 16B
      int idx = t + i * 256;
      int r = idx >> 3, cg = idx & 7;
      async_copy_b128(&As[buf][r * 64 + cg * 8],
                      A + (size_t)(m0 + r) * K + k0 + cg * 8);
    }
#pragma unroll
    for (int i = 0; i < 8; ++i) {  // B: 2048 chunks of 16B
      int idx = t + i * 256;
      int r = idx >> 3, cg = idx & 7;
      async_copy_b128(&Bs[buf][r * 64 + cg * 8],
                      Bt + (size_t)(n0 + r) * K + k0 + cg * 8);
    }
  };

  stage(0, 0);
  const int T = K / 64;
  for (int kt = 0; kt < T; ++kt) {
    const int buf = kt & 1;
    wait_async0();       // this wave's copies for tile kt have landed
    __syncthreads();     // all waves' copies landed; prev reads done
    if (kt + 1 < T) stage((kt + 1) * 64, buf ^ 1);  // overlap with compute

#pragma unroll
    for (int f = 0; f < 2; ++f) {
      v16bf afrag[4];
#pragma unroll
      for (int i = 0; i < 4; ++i)
        afrag[i] = make_afrag(
            &As[buf][(wm * 64 + i * 16 + n16) * 64 + f * 32], half);
      v16bf bfrag[4];
#pragma unroll
      for (int j = 0; j < 4; ++j)
        bfrag[j] =
            *(const v16bf*)(&Bs[buf][(wn * 64 + j * 16 + n16) * 64 + f * 32] +
                            half * 16);
#pragma unroll
      for (int i = 0; i < 4; ++i)
#pragma unroll
        for (int j = 0; j < 4; ++j)
          acc[i][j] = wmma_bf16(afrag[i], bfrag[j], acc[i][j]);
    }
  }

#pragma unroll
  for (int i = 0; i < 4; ++i) {
#pragma unroll
    for (int j = 0; j < 4; ++j) {
      int gmb = m0 + wm * 64 + i * 16 + half * 8;
      int gn = n0 + wn * 64 + j * 16 + n16;
      float* cp = C + (size_t)gmb * N + gn;
#pragma unroll
      for (int r = 0; r < 8; ++r) cp[(size_t)r * N] = acc[i][j][r];
    }
  }
}

// ---------------------------------------------------------------------------
// RoPE + pack: src f32 [B,S,NH,HD] -> dst bf16 [B,NH,S,HD]; gain folds the
// attention 1/sqrt(HD) scale into Q at conversion time.
// ---------------------------------------------------------------------------
__global__ __launch_bounds__(256) void rope_pack_kernel(
    const float* __restrict__ src, const float* __restrict__ fcos,
    const float* __restrict__ fsin, __bf16* __restrict__ dst, int NH,
    float gain) {
  size_t i = (size_t)blockIdx.x * 256 + threadIdx.x;  // B*S*NH*(HD/2)
  int d2 = (int)(i & 63);
  size_t r = i >> 6;
  int hh = (int)(r % NH); r /= NH;
  int s = (int)(r % S_);
  int b = (int)(r / S_);
  size_t si = (((size_t)(b * S_ + s) * NH) + hh) * HD_ + 2 * d2;
  float x0 = src[si], x1 = src[si + 1];
  float c = fcos[s * 64 + d2], sn = fsin[s * 64 + d2];
  size_t di = (((size_t)(b * NH + hh) * S_) + s) * HD_ + 2 * d2;
  dst[di] = (__bf16)((x0 * c - x1 * sn) * gain);
  dst[di + 1] = (__bf16)((x0 * sn + x1 * c) * gain);
}

// pack V transposed: f32 [B,S,KV,HD] -> bf16 [B,KV,HD,S]  (hd-major so the
// attention V tile is a row-contiguous async DMA copy)
__global__ __launch_bounds__(256) void pack_v_kernel(
    const float* __restrict__ src, __bf16* __restrict__ dst) {
  size_t i = (size_t)blockIdx.x * 256 + threadIdx.x;
  int hd = (int)(i % HD_);
  size_t r = i / HD_;
  int kv = (int)(r % KV_); r /= KV_;
  int s = (int)(r % S_);
  int b = (int)(r / S_);
  dst[(((size_t)(b * KV_ + kv) * HD_) + hd) * S_ + s] = (__bf16)src[i];
}

// ---------------------------------------------------------------------------
// Flash attention: 128 threads = 4 waves; wave w owns q rows
// [qb*64 + w*16, +16) of one (b,h). Causal, online softmax. Q pre-scaled.
// Q bf16 [B,H,S,HD]; K bf16 [B,KV,S,HD]; Vt bf16 [B,KV,HD,S]; O bf16 [B,S,H,HD].
// K and V tiles both staged by double-buffered async DMA.
// ---------------------------------------------------------------------------
__global__ __launch_bounds__(128) void attn_kernel(
    const __bf16* __restrict__ Q, const __bf16* __restrict__ Kc,
    const __bf16* __restrict__ Vt, __bf16* __restrict__ O) {
  __shared__ __bf16 Ks[2][32 * 128];  // [key][hd]
  __shared__ __bf16 Vs[2][128 * 32];  // [hd][key]
  __shared__ __bf16 Ps[4 * 16 * 32];  // per-wave probs scratch

  const int t = threadIdx.x;
  const int lane = t & 31;
  const int wave = t >> 5;
  const int half = lane >> 4;
  const int n16 = lane & 15;

  int blk = blockIdx.x;  // ((b*H + h)*(S/64) + qb)
  int qb = blk & 31; blk >>= 5;
  int h = blk & 31;
  int b = blk >> 5;
  int kvh = h / NREP_;
  int q0 = qb * 64 + wave * 16;

  const __bf16* qbase = Q + (size_t)(b * H_ + h) * S_ * HD_;
  const __bf16* kbase = Kc + (size_t)(b * KV_ + kvh) * S_ * HD_;
  const __bf16* vbase = Vt + (size_t)(b * KV_ + kvh) * HD_ * S_;

  v16bf qf[4];
  {
    const __bf16* qrow = qbase + (size_t)(q0 + n16) * HD_;
#pragma unroll
    for (int f = 0; f < 4; ++f) qf[f] = make_afrag(qrow + f * 32, half);
  }

  v8f oacc[8];
#pragma unroll
  for (int i = 0; i < 8; ++i) oacc[i] = (v8f){0, 0, 0, 0, 0, 0, 0, 0};
  float mrun[8], lrun[8];
#pragma unroll
  for (int r = 0; r < 8; ++r) { mrun[r] = -1e30f; lrun[r] = 0.f; }

  auto stageKV = [&](int kk, int buf) {
#pragma unroll
    for (int i = 0; i < 4; ++i) {
      int idx = t + i * 128;  // 0..511 chunks of 16B each side
      int row = idx >> 4, g = idx & 15;
      async_copy_b128(&Ks[buf][row * 128 + g * 8],
                      kbase + (size_t)(kk + row) * HD_ + g * 8);
      int hd = idx >> 2, c = idx & 3;
      async_copy_b128(&Vs[buf][hd * 32 + c * 8],
                      vbase + (size_t)hd * S_ + kk + c * 8);
    }
  };

  const int ktiles = qb * 2 + 2;
  stageKV(0, 0);
  for (int kt = 0; kt < ktiles; ++kt) {
    const int kk = kt * 32;
    const int buf = kt & 1;
    wait_async0();
    __syncthreads();
    if (kt + 1 < ktiles) stageKV(kk + 32, buf ^ 1);

    v8f sacc[2];
    sacc[0] = (v8f){0, 0, 0, 0, 0, 0, 0, 0};
    sacc[1] = (v8f){0, 0, 0, 0, 0, 0, 0, 0};
#pragma unroll
    for (int nt = 0; nt < 2; ++nt) {
      const __bf16* col = &Ks[buf][(nt * 16 + n16) * 128];
#pragma unroll
      for (int f = 0; f < 4; ++f) {
        v16bf bfrag = *(const v16bf*)(col + f * 32 + half * 16);
        sacc[nt] = wmma_bf16(qf[f], bfrag, sacc[nt]);
      }
    }

    float pr0[8], pr1[8];
#pragma unroll
    for (int r = 0; r < 8; ++r) {
      int qrow = q0 + half * 8 + r;
      float s0 = sacc[0][r] + ((kk + n16) > qrow ? -1e30f : 0.f);
      float s1 = sacc[1][r] + ((kk + 16 + n16) > qrow ? -1e30f : 0.f);
      float m = fmaxf(s0, s1);
#pragma unroll
      for (int off = 1; off < 16; off <<= 1)
        m = fmaxf(m, __shfl_xor(m, off, 32));
      float mnew = fmaxf(mrun[r], m);
      float alpha = __expf(mrun[r] - mnew);
      mrun[r] = mnew;
      float p0 = __expf(s0 - mnew);
      float p1 = __expf(s1 - mnew);
      pr0[r] = p0; pr1[r] = p1;
      float rs = p0 + p1;
#pragma unroll
      for (int off = 1; off < 16; off <<= 1) rs += __shfl_xor(rs, off, 32);
      lrun[r] = lrun[r] * alpha + rs;
#pragma unroll
      for (int c8 = 0; c8 < 8; ++c8) oacc[c8][r] *= alpha;
    }

    __bf16* pw = &Ps[wave * 16 * 32];
#pragma unroll
    for (int r = 0; r < 8; ++r) {
      pw[(half * 8 + r) * 32 + n16] = (__bf16)pr0[r];
      pw[(half * 8 + r) * 32 + 16 + n16] = (__bf16)pr1[r];
    }
    asm volatile("s_wait_dscnt 0" ::: "memory");
    v16bf pf = make_afrag(pw + n16 * 32, half);

#pragma unroll
    for (int c8 = 0; c8 < 8; ++c8) {
      v16bf vfrag = *(const v16bf*)(&Vs[buf][(c8 * 16 + n16) * 32] + half * 16);
      oacc[c8] = wmma_bf16(pf, vfrag, oacc[c8]);
    }
  }

  // epilogue: normalize, store bf16 O[b][qrow][h][hd]
#pragma unroll
  for (int c8 = 0; c8 < 8; ++c8) {
#pragma unroll
    for (int r = 0; r < 8; ++r) {
      int qrow = q0 + half * 8 + r;
      float val = oacc[c8][r] / lrun[r];
      O[(((size_t)b * S_ + qrow) * H_ + h) * HD_ + c8 * 16 + n16] =
          (__bf16)val;
    }
  }
}

// ---------------------------------------------------------------------------
extern "C" void kernel_launch(void* const* d_in, const int* in_sizes, int n_in,
                              void* d_out, int out_size, void* d_ws,
                              size_t ws_size, hipStream_t stream) {
  (void)in_sizes; (void)n_in; (void)out_size; (void)ws_size;
  const float* x  = (const float*)d_in[0];
  const float* wq = (const float*)d_in[1];
  const float* wk = (const float*)d_in[2];
  const float* wv = (const float*)d_in[3];
  const float* wo = (const float*)d_in[4];
  const float* fcos = (const float*)d_in[7];
  const float* fsin = (const float*)d_in[8];
  float* out = (float*)d_out;

  const size_t XN = (size_t)B_ * S_ * D_;         // 33,554,432
  const size_t QN = (size_t)B_ * S_ * H_ * HD_;   // 33,554,432
  const size_t KN = (size_t)B_ * S_ * KV_ * HD_;  //  8,388,608
  const size_t WQ = (size_t)D_ * H_ * HD_;        // 16,777,216
  const size_t WK = (size_t)D_ * KV_ * HD_;       //  4,194,304

  __bf16* xb  = (__bf16*)d_ws;
  __bf16* wqT = xb + XN;
  __bf16* wkT = wqT + WQ;
  __bf16* wvT = wkT + WK;
  __bf16* woT = wvT + WK;
  float*  xq  = (float*)(woT + WQ);
  float*  xk  = xq + QN;
  float*  xv  = xk + KN;
  __bf16* qb  = (__bf16*)(xv + KN);
  __bf16* kb  = qb + QN;
  __bf16* vb  = kb + KN;
  __bf16* attnb = vb + KN;

  const int M = B_ * S_;  // 8192

  // one-time precision/layout pre-pass
  conv_bf16_kernel<<<(int)(XN / 8 / 256), 256, 0, stream>>>(x, xb);
  transpose_conv_kernel<4096, 4096><<<dim3(128, 128), 256, 0, stream>>>(wq, wqT);
  transpose_conv_kernel<1024, 4096><<<dim3(32, 128), 256, 0, stream>>>(wk, wkT);
  transpose_conv_kernel<1024, 4096><<<dim3(32, 128), 256, 0, stream>>>(wv, wvT);
  transpose_conv_kernel<4096, 4096><<<dim3(128, 128), 256, 0, stream>>>(wo, woT);

  // QKV projections (double-buffered async-staged WMMA GEMMs)
  gemm_bf16_kernel<4096, 4096><<<dim3(16, 64), 256, 0, stream>>>(xb, wqT, xq, M);
  gemm_bf16_kernel<1024, 4096><<<dim3(4, 64), 256, 0, stream>>>(xb, wkT, xk, M);
  gemm_bf16_kernel<1024, 4096><<<dim3(4, 64), 256, 0, stream>>>(xb, wvT, xv, M);

  // RoPE + pack (Q pre-scaled by 1/sqrt(HD))
  rope_pack_kernel<<<(int)(QN / 2 / 256), 256, 0, stream>>>(xq, fcos, fsin, qb,
                                                            H_, SCALE_);
  rope_pack_kernel<<<(int)(KN / 2 / 256), 256, 0, stream>>>(xk, fcos, fsin, kb,
                                                            KV_, 1.0f);
  pack_v_kernel<<<(int)(KN / 256), 256, 0, stream>>>(xv, vb);

  // Flash attention
  attn_kernel<<<B_ * H_ * (S_ / 64), 128, 0, stream>>>(qb, kb, vb, attnb);

  // Output projection
  gemm_bf16_kernel<4096, 4096><<<dim3(16, 64), 256, 0, stream>>>(attnb, woT, out, M);
}
